// Adaptive_Fusion_4535485464591
// MI455X (gfx1250) — compile-verified, matrix operand order checked by reading, not run on hardware
//
#include <hip/hip_runtime.h>
#include <hip/hip_bf16.h>

typedef __attribute__((ext_vector_type(16))) _Float16     v16h;
typedef __attribute__((ext_vector_type(8)))  float        v8f;
typedef __attribute__((ext_vector_type(4)))  unsigned int v4u;
typedef __attribute__((ext_vector_type(8)))  int          v8i;
typedef __attribute__((ext_vector_type(4)))  int          v4i;

#define BDIM   8
#define TDIM   64
#define NDIM   256
#define HDIM   8
#define FEAT   512
#define MTOK   (BDIM * TDIM * NDIM)   // 131072 token rows

// ---------------------------------------------------------------------------
// Tensor Data Mover: 2-D tile load Global -> LDS.
// D# built per CDNA5 ISA ch.8: group0 {count=1, lds_addr, global_addr, type=2},
// group1 {data_size=2B, tensor_dim0/1, tile_dim0/1, tensor_dim0_stride}.
// ROCm7.2 clang-22 builtin is 5-arg, clang-23 is 6-arg.
// ---------------------------------------------------------------------------
__device__ __forceinline__ void tdm_raw(v4u g0, v8i g1, v4i g2, v4i g3) {
#if defined(__clang_major__) && (__clang_major__ >= 23)
  v8i g4 = {0, 0, 0, 0, 0, 0, 0, 0};
  __builtin_amdgcn_tensor_load_to_lds(g0, g1, g2, g3, g4, 0);
#else
  __builtin_amdgcn_tensor_load_to_lds(g0, g1, g2, g3, 0);
#endif
}

__device__ __forceinline__ void tdm_load_2d(unsigned ldsOff, const void* g,
                                            unsigned tileK, unsigned tileRows,
                                            unsigned td0, unsigned td1,
                                            unsigned long long strideElems) {
  unsigned long long ga = (unsigned long long)(size_t)g;
  v4u g0 = {1u,                                   // count=1, user descriptor
            ldsOff,                               // lds byte address
            (unsigned)ga,                         // global_addr[31:0]
            (unsigned)((ga >> 32) & 0x01ffffffu) | 0x80000000u};  // addr[56:32]|type=2
  v8i g1 = {(int)(1u << 16),                      // wg_mask=0, data_size=1 (2 bytes)
            (int)((td0 & 0xffffu) << 16),         // tensor_dim0[15:0] @ bits63:48
            (int)(((td0 >> 16) & 0xffffu) | ((td1 & 0xffffu) << 16)),
            (int)(((td1 >> 16) & 0xffffu) | ((tileK & 0xffffu) << 16)),  // tile_dim0
            (int)(tileRows & 0xffffu),            // tile_dim1, tile_dim2=0
            (int)(unsigned)(strideElems & 0xffffffffu),   // dim0_stride[31:0]
            (int)((strideElems >> 32) & 0xffffu),         // dim0_stride[47:32]
            0};
  v4i z = {0, 0, 0, 0};
  tdm_raw(g0, g1, z, z);
}

// ---------------------------------------------------------------------------
// Fragment loaders matching the CDNA5 16-bit WMMA VGPR layout (ISA 7.12.2):
// lane 0-15 holds K = {0..7, 16..23}, lane 16-31 holds K = {8..15, 24..31}.
// ---------------------------------------------------------------------------
__device__ __forceinline__ v16h frag_ld(const _Float16* base, int row, int stride,
                                        int k0, int hi) {
  v16h f;
  const _Float16* p = base + row * stride + k0 + hi * 8;
#pragma unroll
  for (int j = 0; j < 4; ++j) {
    f[2 * j]     = p[2 * j];
    f[2 * j + 1] = p[2 * j + 1];
    f[8 + 2 * j] = p[16 + 2 * j];
    f[9 + 2 * j] = p[16 + 2 * j + 1];
  }
  return f;
}

// B fragment of V^T where V is stored row-major [s][64] in LDS.
__device__ __forceinline__ v16h frag_ld_T(const _Float16* base, int col,
                                          int k0, int hi) {
  v16h f;
  const int o = k0 + hi * 8;
#pragma unroll
  for (int j = 0; j < 4; ++j) {
    f[2 * j]     = base[(o + 2 * j) * 64 + col];
    f[2 * j + 1] = base[(o + 2 * j + 1) * 64 + col];
    f[8 + 2 * j] = base[(o + 16 + 2 * j) * 64 + col];
    f[9 + 2 * j] = base[(o + 16 + 2 * j + 1) * 64 + col];
  }
  return f;
}

// ---------------------------------------------------------------------------
// 1) residual adds + f16 downconvert
// ---------------------------------------------------------------------------
__global__ __launch_bounds__(256) void prep_kernel(
    const float* __restrict__ xl, const float* __restrict__ xh,
    const float* __restrict__ te, float* __restrict__ xlteF,
    _Float16* __restrict__ xlteH, _Float16* __restrict__ xhteH) {
  long i0 = ((long)blockIdx.x * 256 + threadIdx.x) * 8;
#pragma unroll
  for (int j = 0; j < 8; ++j) {
    long i = i0 + j;
    float t = te[i];
    float a = xl[i] + t;
    float c = xh[i] + t;
    xlteF[i] = a;
    xlteH[i] = (_Float16)a;
    xhteH[i] = (_Float16)c;
  }
}

// ---------------------------------------------------------------------------
// 2) weight convert+transpose: Wt[n][k] = (f16) W[k][n]
// ---------------------------------------------------------------------------
__global__ __launch_bounds__(256) void cvt_w_kernel(const float* __restrict__ W,
                                                    _Float16* __restrict__ Wt) {
  int i = blockIdx.x * 256 + threadIdx.x;   // i = n*512 + k
  int k = i & 511, n = i >> 9;
  Wt[i] = (_Float16)W[k * 512 + n];
}

// ---------------------------------------------------------------------------
// 3) WMMA GEMM: block = 64 rows x FULL 512 cols (A read from HBM exactly once).
//    8 waves; wave w owns cols [w*64, w*64+64). 16 WMMAs / wave / K-step.
//    A (64x32) and B (512x32) slabs staged to LDS by the Tensor Data Mover.
// ---------------------------------------------------------------------------
template <bool RELU, bool F16OUT>
__global__ __launch_bounds__(256) void gemm512(
    const _Float16* __restrict__ A, const _Float16* __restrict__ Wt,
    const float* __restrict__ bias, _Float16* __restrict__ outH,
    float* __restrict__ outF) {
  __shared__ _Float16 smem[2048 + 512 * 32];   // A slab 4KB @0, B slab 32KB @4096B
  _Float16* As = smem;
  _Float16* Bs = smem + 2048;

  const int tid  = threadIdx.x;
  const int lane = tid & 31;
  const int w    = tid >> 5;
  const int hi   = lane >> 4, r = lane & 15;
  const long mBase = (long)blockIdx.x * 64;

  v8f acc[16];
#pragma unroll
  for (int i = 0; i < 16; ++i) acc[i] = (v8f){0, 0, 0, 0, 0, 0, 0, 0};

#pragma unroll 1
  for (int kk = 0; kk < 512; kk += 32) {
    if (tid < 32) {  // wave 0 drives the TDM, waits TENSORcnt, then barrier
      tdm_load_2d(0u,    A + mBase * 512 + kk, 32, 64, 512, (unsigned)MTOK, 512ull);
      tdm_load_2d(4096u, Wt + kk,              32, 512, 512, 512, 512ull);
      __builtin_amdgcn_s_wait_tensorcnt(0);
    }
    __syncthreads();

    v16h bf[4];
#pragma unroll
    for (int ni = 0; ni < 4; ++ni)
      bf[ni] = frag_ld(Bs, w * 64 + ni * 16 + r, 32, 0, hi);
#pragma unroll
    for (int mi = 0; mi < 4; ++mi) {
      v16h af = frag_ld(As, mi * 16 + r, 32, 0, hi);
#pragma unroll
      for (int ni = 0; ni < 4; ++ni)
        acc[mi * 4 + ni] = __builtin_amdgcn_wmma_f32_16x16x32_f16(
            false, af, false, bf[ni], (short)0, acc[mi * 4 + ni], false, false);
    }
    __syncthreads();  // readers done before next TDM overwrites the slabs
  }

#pragma unroll
  for (int mi = 0; mi < 4; ++mi) {
#pragma unroll
    for (int ni = 0; ni < 4; ++ni) {
      int col  = w * 64 + ni * 16 + r;
      float bv = bias[col];
      const v8f& a = acc[mi * 4 + ni];
#pragma unroll
      for (int j = 0; j < 8; ++j) {
        long m  = mBase + mi * 16 + j + hi * 8;
        float v = a[j] + bv;
        if constexpr (RELU) v = fmaxf(v, 0.0f);
        if constexpr (F16OUT) outH[m * 512 + col] = (_Float16)v;
        else                  outF[m * 512 + col] = v;
      }
    }
  }
}

// ---------------------------------------------------------------------------
// 4) fused causal attention head: one block per (b,n,h).
//    Q/K/V 64x64 tiles fetched by TDM (handles the t-stride of 256*512 elems),
//    S = QK^T (WMMA), mask-before-scale softmax, O = PV (WMMA).
// ---------------------------------------------------------------------------
__global__ __launch_bounds__(256) void attn_kernel(
    const _Float16* __restrict__ Qg, const _Float16* __restrict__ Kg,
    const _Float16* __restrict__ Vg, _Float16* __restrict__ Og) {
  __shared__ __align__(16) unsigned char smemRaw[24576 + 16384];
  _Float16* Qs = (_Float16*)smemRaw;               // 8KB, reused for P
  _Float16* Ks = (_Float16*)(smemRaw + 8192);      // 8KB
  _Float16* Vs = (_Float16*)(smemRaw + 16384);     // 8KB
  float*    Ss = (float*)(smemRaw + 24576);        // 16KB

  const int bid = blockIdx.x;
  const int h = bid & 7;
  const int n = (bid >> 3) & 255;
  const int b = bid >> 11;
  const long tokBase = (long)b * TDIM * NDIM + n;  // token(t) = tokBase + t*256
  const long gBase   = tokBase * 512 + h * 64;     // addr(t,d) = gBase + t*131072 + d
  const int tid = threadIdx.x;

  if (tid < 32) {
    const unsigned long long st = (unsigned long long)NDIM * 512;  // t-stride (elems)
    tdm_load_2d(0u,     Qg + gBase, 64, 64, 1u << 30, 1u << 30, st);
    tdm_load_2d(8192u,  Kg + gBase, 64, 64, 1u << 30, 1u << 30, st);
    tdm_load_2d(16384u, Vg + gBase, 64, 64, 1u << 30, 1u << 30, st);
    __builtin_amdgcn_s_wait_tensorcnt(0);
  }
  __syncthreads();

  const int lane = tid & 31, w = tid >> 5;
  const int wm = w & 3, wn = w >> 2;
  const int hi = lane >> 4, r = lane & 15;

  // S = Q K^T  (M=64 t, N=64 s, K=64 d)
  v8f s0 = {}, s1 = {};
#pragma unroll
  for (int k0 = 0; k0 < 64; k0 += 32) {
    v16h aq = frag_ld(Qs, wm * 16 + r, 64, k0, hi);
    v16h b0 = frag_ld(Ks, wn * 32 + r, 64, k0, hi);
    v16h b1 = frag_ld(Ks, wn * 32 + 16 + r, 64, k0, hi);
    s0 = __builtin_amdgcn_wmma_f32_16x16x32_f16(false, aq, false, b0,
                                                (short)0, s0, false, false);
    s1 = __builtin_amdgcn_wmma_f32_16x16x32_f16(false, aq, false, b1,
                                                (short)0, s1, false, false);
  }
#pragma unroll
  for (int f = 0; f < 2; ++f) {
    const v8f& s = f ? s1 : s0;
    int col = wn * 32 + f * 16 + r;
#pragma unroll
    for (int j = 0; j < 8; ++j) Ss[(wm * 16 + j + hi * 8) * 64 + col] = s[j];
  }
  __syncthreads();

  // softmax per row t (mask BEFORE scale, like reference: masked logit = -32767/8)
  if (tid < 64) {
    const int t = tid;
    float mx = -3.4e38f;
    for (int s = 0; s <= t; ++s) mx = fmaxf(mx, Ss[t * 64 + s] * 0.125f);
    float sum = 0.0f;
    for (int s = 0; s < 64; ++s) {
      float x = (s <= t) ? Ss[t * 64 + s] * 0.125f : -4095.875f;
      float e = __expf(x - mx);
      sum += e;
      Ss[t * 64 + s] = e;
    }
    float inv = 1.0f / sum;
    for (int s = 0; s < 64; ++s) Qs[t * 64 + s] = (_Float16)(Ss[t * 64 + s] * inv);
  }
  __syncthreads();

  // O = P V  (M=64 t, N=64 d, K=64 s); B fragment = V^T gathered from LDS
  v8f o0 = {}, o1 = {};
#pragma unroll
  for (int k0 = 0; k0 < 64; k0 += 32) {
    v16h ap = frag_ld(Qs, wm * 16 + r, 64, k0, hi);
    v16h b0 = frag_ld_T(Vs, wn * 32 + r, k0, hi);
    v16h b1 = frag_ld_T(Vs, wn * 32 + 16 + r, k0, hi);
    o0 = __builtin_amdgcn_wmma_f32_16x16x32_f16(false, ap, false, b0,
                                                (short)0, o0, false, false);
    o1 = __builtin_amdgcn_wmma_f32_16x16x32_f16(false, ap, false, b1,
                                                (short)0, o1, false, false);
  }
#pragma unroll
  for (int f = 0; f < 2; ++f) {
    const v8f& o = f ? o1 : o0;
    int d = wn * 32 + f * 16 + r;
#pragma unroll
    for (int j = 0; j < 8; ++j) {
      int t = wm * 16 + j + hi * 8;
      Og[gBase + (long)t * (NDIM * 512) + d] = (_Float16)o[j];
    }
  }
}

// ---------------------------------------------------------------------------
// 5) residual + LayerNorm (eps 1e-5, no affine): one wave per token row
// ---------------------------------------------------------------------------
__global__ __launch_bounds__(256) void add_ln_kernel(
    const float* __restrict__ x, const float* __restrict__ res,
    float* __restrict__ outF, _Float16* __restrict__ outH) {
  const int w = threadIdx.x >> 5, lane = threadIdx.x & 31;
  const long tok = (long)blockIdx.x * 8 + w;
  const float* px = x + tok * 512;
  const float* pr = res + tok * 512;
  float v[16];
  float sum = 0.0f;
#pragma unroll
  for (int i = 0; i < 16; ++i) {
    float t = px[i * 32 + lane] + pr[i * 32 + lane];
    v[i] = t;
    sum += t;
  }
#pragma unroll
  for (int off = 16; off > 0; off >>= 1) sum += __shfl_xor(sum, off, 32);
  float mean = sum * (1.0f / 512.0f);
  float var = 0.0f;
#pragma unroll
  for (int i = 0; i < 16; ++i) { float d = v[i] - mean; var += d * d; }
#pragma unroll
  for (int off = 16; off > 0; off >>= 1) var += __shfl_xor(var, off, 32);
  float inv = rsqrtf(var * (1.0f / 512.0f) + 1e-5f);
#pragma unroll
  for (int i = 0; i < 16; ++i) {
    float y = (v[i] - mean) * inv;
    outF[tok * 512 + i * 32 + lane] = y;
    if (outH) outH[tok * 512 + i * 32 + lane] = (_Float16)y;
  }
}

// ---------------------------------------------------------------------------
extern "C" void kernel_launch(void* const* d_in, const int* in_sizes, int n_in,
                              void* d_out, int out_size, void* d_ws, size_t ws_size,
                              hipStream_t stream) {
  (void)in_sizes; (void)n_in; (void)out_size; (void)ws_size;
  const float* xl = (const float*)d_in[0];
  const float* xh = (const float*)d_in[1];
  const float* te = (const float*)d_in[2];
  const float* Wq = (const float*)d_in[3];  const float* bq = (const float*)d_in[4];
  const float* Wk = (const float*)d_in[5];  const float* bk = (const float*)d_in[6];
  const float* Wv = (const float*)d_in[7];  const float* bv = (const float*)d_in[8];
  const float* Wo = (const float*)d_in[9];  const float* bo = (const float*)d_in[10];
  const float* W1 = (const float*)d_in[11]; const float* b1 = (const float*)d_in[12];
  const float* W2 = (const float*)d_in[13]; const float* b2 = (const float*)d_in[14];

  const long M = MTOK;
  const long E = M * 512;

  char* p = (char*)d_ws;
  auto take = [&](size_t bytes) {
    char* r = p;
    p += (bytes + 255) & ~(size_t)255;
    return r;
  };
  float*    xlteF = (float*)take(E * 4);
  _Float16* xlteH = (_Float16*)take(E * 2);
  _Float16* xhteH = (_Float16*)take(E * 2);
  _Float16* Wt    = (_Float16*)take((size_t)6 * 512 * 512 * 2);
  _Float16* qH    = (_Float16*)take(E * 2);
  _Float16* kH    = (_Float16*)take(E * 2);
  _Float16* vH    = (_Float16*)take(E * 2);
  _Float16* aH    = (_Float16*)take(E * 2);
  _Float16* valH  = (_Float16*)take(E * 2);
  _Float16* h1H   = (_Float16*)take(E * 2);
  float*    tmpF  = (float*)take(E * 4);
  float*    valF  = (float*)take(E * 4);

  prep_kernel<<<dim3((unsigned)(E / 2048)), 256, 0, stream>>>(xl, xh, te, xlteF,
                                                              xlteH, xhteH);
  const float* Wsrc[6] = {Wq, Wk, Wv, Wo, W1, W2};
  for (int i = 0; i < 6; ++i)
    cvt_w_kernel<<<dim3(1024), 256, 0, stream>>>(Wsrc[i], Wt + (size_t)i * 262144);

  dim3 gg((unsigned)(M / 64));
  gemm512<false, true><<<gg, 256, 0, stream>>>(xlteH, Wt + 0 * 262144, bq, qH, nullptr);
  gemm512<true,  true><<<gg, 256, 0, stream>>>(xhteH, Wt + 1 * 262144, bk, kH, nullptr);
  gemm512<true,  true><<<gg, 256, 0, stream>>>(xhteH, Wt + 2 * 262144, bv, vH, nullptr);

  attn_kernel<<<dim3(BDIM * NDIM * HDIM), 256, 0, stream>>>(qH, kH, vH, aH);

  gemm512<false, false><<<gg, 256, 0, stream>>>(aH, Wt + 3 * 262144, bo, nullptr, tmpF);
  add_ln_kernel<<<dim3((unsigned)(M / 8)), 256, 0, stream>>>(tmpF, xlteF, valF, valH);
  gemm512<true,  true><<<gg, 256, 0, stream>>>(valH, Wt + 4 * 262144, b1, h1H, nullptr);
  gemm512<false, false><<<gg, 256, 0, stream>>>(h1H, Wt + 5 * 262144, b2, nullptr, tmpF);
  add_ln_kernel<<<dim3((unsigned)(M / 8)), 256, 0, stream>>>(tmpF, valF, (float*)d_out,
                                                             nullptr);
}
